// SwinTransformerBlock_29695403884786
// MI455X (gfx1250) — compile-verified, matrix operand order checked by reading, not run on hardware
//
#include <hip/hip_runtime.h>
#include <hip/hip_bf16.h>
#include <math.h>

// ---------------------------------------------------------------------------
// Swin block for MI455X (gfx1250): all GEMMs via v_wmma_f32_16x16x32_bf16.
// GEMMs: 128x64 block tile, 8 waves, each wave a 16x64 strip (4 accumulators,
// A-fragment reused across 4 WMMAs). B panels staged to LDS with gfx1250
// async global->LDS (ASYNCcnt); MLP2 double-buffers panels so the async DMA
// of chunk i+1 overlaps WMMA compute of chunk i.
// ---------------------------------------------------------------------------

typedef __attribute__((ext_vector_type(16))) __bf16 v16bf;
typedef __attribute__((ext_vector_type(8)))  __bf16 v8bf;
typedef __attribute__((ext_vector_type(8)))  float  v8f;
typedef __attribute__((ext_vector_type(4)))  int    v4i;

static constexpr int kB    = 16;
static constexpr int kHW   = 56;        // H = W = 56
static constexpr int kC    = 192;
static constexpr int kL    = kHW * kHW; // 3136
static constexpr int kM    = kB * kL;   // 50176 rows
static constexpr int kNH   = 6;
static constexpr int kHD   = 32;
static constexpr int kNP   = 64;        // window tokens padded 49 -> 64
static constexpr int kHID  = 768;
static constexpr int kSH   = 3;         // shift
static constexpr int kKC   = 192;       // K chunk staged in LDS per iteration
static constexpr int kPanel = 64 * (kKC + 8);   // LDS elements per panel

#if __has_builtin(__builtin_amdgcn_global_load_async_to_lds_b128)
#define HAS_ASYNC_LDS 1
#endif

#define AS_GLOBAL __attribute__((address_space(1)))
#define AS_LOCAL  __attribute__((address_space(3)))

__device__ __forceinline__ void async_wait_zero() {
#if __has_builtin(__builtin_amdgcn_s_wait_asynccnt)
  __builtin_amdgcn_s_wait_asynccnt(0);
#else
  asm volatile("s_wait_asynccnt 0x0" ::: "memory");
#endif
}

// ---------------------------------------------------------------------------
__device__ __forceinline__ float wave_sum(float v) {
#pragma unroll
  for (int m = 16; m >= 1; m >>= 1) v += __shfl_xor(v, m, 32);
  return v;
}

// Load one lane's share of a 16-bit 16x32 WMMA A/B fragment.
// ISA layout (CDNA5 ISA): lanes 0-15 hold K = [0..7] and [16..23],
// lanes 16-31 hold K = [8..15] and [24..31] of their row/column.
__device__ __forceinline__ v16bf load_frag(const __bf16* row, int k0, int half) {
  const __bf16* p = row + k0 + half * 8;
  v8bf lo = *(const v8bf*)(p);
  v8bf hi = *(const v8bf*)(p + 16);
  v16bf r;
#pragma unroll
  for (int i = 0; i < 8; ++i) { r[i] = lo[i]; r[i + 8] = hi[i]; }
  return r;
}

// Issue the copy of a 64-column x KC-deep B panel (column data K-contiguous,
// row stride ldb) into LDS with +8 element row padding. Async (ASYNCcnt) when
// available; caller is responsible for stage_B_drain() + barrier.
template <int KC>
__device__ __forceinline__ void stage_B_issue(const __bf16* __restrict__ Bg,
                                              int ldb, __bf16* Bs) {
  constexpr int CPR = KC / 8;       // 16B chunks per row
  constexpr int TOT = 64 * CPR;     // multiple of 256 for KC = 192
  for (int c = threadIdx.x; c < TOT; c += 256) {
    const int row = c / CPR, col = (c % CPR) * 8;
    const __bf16* src = Bg + (size_t)row * ldb + col;
    __bf16* dst = Bs + row * (KC + 8) + col;
#ifdef HAS_ASYNC_LDS
    __builtin_amdgcn_global_load_async_to_lds_b128(
        (AS_GLOBAL v4i*)(AS_GLOBAL void*)src,
        (AS_LOCAL v4i*)(AS_LOCAL void*)dst, 0, 0);
#else
    *(v8bf*)dst = *(const v8bf*)src;
#endif
  }
}

__device__ __forceinline__ void stage_B_drain() {
#ifdef HAS_ASYNC_LDS
  async_wait_zero();
#endif
}

// Accumulate a 16x64 strip over one KC-deep chunk: one A fragment per k-step
// feeds 4 WMMAs. All 4 B fragments are loaded into distinct registers first
// so the scheduler can overlap LDS traffic with WMMA issue (partial DScnt
// waits instead of a dscnt==0 drain per WMMA).
template <int KC>
__device__ __forceinline__ void gemm_chunk(const __bf16* arow, const __bf16* Bs,
                                           int lane, v8f acc[4]) {
  const int half = lane >> 4, lm = lane & 15;
#pragma unroll
  for (int k0 = 0; k0 < KC; k0 += 32) {
    v16bf a = load_frag(arow, k0, half);
    v16bf b[4];
#pragma unroll
    for (int j = 0; j < 4; ++j)
      b[j] = load_frag(Bs + (j * 16 + lm) * (KC + 8), k0, half);
#pragma unroll
    for (int j = 0; j < 4; ++j)
      acc[j] = __builtin_amdgcn_wmma_f32_16x16x32_bf16(false, a, false, b[j],
                                                       (short)0, acc[j],
                                                       false, false);
  }
}

// Direct (global/LDS A and B, no staging) 16x16 tile used by attention.
template <int KTOT>
__device__ __forceinline__ v8f wmma_tile(const __bf16* A, int lda,
                                         const __bf16* B, int ldb, int lane) {
  const int half = lane >> 4, lm = lane & 15;
  const __bf16* arow = A + (size_t)lm * lda;
  const __bf16* brow = B + (size_t)lm * ldb;
  v8f acc = {};
#pragma unroll
  for (int k0 = 0; k0 < KTOT; k0 += 32) {
    v16bf a = load_frag(arow, k0, half);
    v16bf b = load_frag(brow, k0, half);
    acc = __builtin_amdgcn_wmma_f32_16x16x32_bf16(false, a, false, b,
                                                  (short)0, acc, false, false);
  }
  return acc;
}

// ---------------------------------------------------------------------------
// Kernel 1: convert fp32 weights to bf16 ([out, in] row-major kept, so every
// WMMA B-fragment column is K-contiguous).
__global__ void prep_weights(const float* __restrict__ qkvw,
                             const float* __restrict__ m1w,
                             const float* __restrict__ m2w,
                             __bf16* __restrict__ wq, __bf16* __restrict__ w1,
                             __bf16* __restrict__ w2) {
  int i = blockIdx.x * blockDim.x + threadIdx.x;
  if (i < 3 * kC * kC) wq[i] = (__bf16)qkvw[i];
  if (i < kHID * kC)   w1[i] = (__bf16)m1w[i];
  if (i < kC * kHID)   w2[i] = (__bf16)m2w[i];
}

// ---------------------------------------------------------------------------
// Kernel 2: LN1 -> shift(-3,-3) + window partition gather -> attn LN -> bf16.
__global__ void stage1_ln_window(const float* __restrict__ x,
                                 const float* __restrict__ g1, const float* __restrict__ b1,
                                 const float* __restrict__ g2, const float* __restrict__ b2,
                                 __bf16* __restrict__ Awin) {
  const int row = (int)((blockIdx.x * blockDim.x + threadIdx.x) >> 5);
  const int lane = threadIdx.x & 31;
  if (row >= kM) return;
  const int w = row / 49, t = row % 49;
  const int b = w >> 6, wi = w & 63, wh = wi >> 3, ww = wi & 7;
  const int i = t / 7, j = t % 7;
  const int hs = (wh * 7 + i + kSH) % kHW;
  const int vs = (ww * 7 + j + kSH) % kHW;
  const float* src = x + ((size_t)b * kL + hs * kHW + vs) * kC;

  float v[6], s = 0.f, s2 = 0.f;
#pragma unroll
  for (int k = 0; k < 6; ++k) {
    v[k] = src[k * 32 + lane];
    s += v[k]; s2 += v[k] * v[k];
  }
  s = wave_sum(s); s2 = wave_sum(s2);
  float mean = s * (1.f / kC);
  float inv  = rsqrtf(s2 * (1.f / kC) - mean * mean + 1e-5f);

  float y[6], t1 = 0.f, t2 = 0.f;
#pragma unroll
  for (int k = 0; k < 6; ++k) {
    int c = k * 32 + lane;
    y[k] = (v[k] - mean) * inv * g1[c] + b1[c];
    t1 += y[k]; t2 += y[k] * y[k];
  }
  t1 = wave_sum(t1); t2 = wave_sum(t2);
  float m2   = t1 * (1.f / kC);
  float inv2 = rsqrtf(t2 * (1.f / kC) - m2 * m2 + 1e-5f);

  __bf16* dst = Awin + (size_t)row * kC;
#pragma unroll
  for (int k = 0; k < 6; ++k) {
    int c = k * 32 + lane;
    dst[c] = (__bf16)((y[k] - m2) * inv2 * g2[c] + b2[c]);
  }
}

// ---------------------------------------------------------------------------
// Kernel 3: QKV GEMM [50176 x 192] x [192 x 576]; epilogue scatters Q,K into
// [win,head,64(pad),32] and V transposed into [win,head,32,64(pad)].
__global__ void qkv_gemm(const __bf16* __restrict__ A, const __bf16* __restrict__ W,
                         const float* __restrict__ bias,
                         __bf16* __restrict__ qb, __bf16* __restrict__ kb,
                         __bf16* __restrict__ vt) {
  extern __shared__ __bf16 Bs[];
  const int lane = threadIdx.x & 31, wv = threadIdx.x >> 5;
  const int half = lane >> 4, lm = lane & 15;
  const int m0 = blockIdx.x * 128 + wv * 16;
  const int nb = blockIdx.y * 64;

  stage_B_issue<kKC>(W + (size_t)nb * kC, kC, Bs);
  stage_B_drain();
  __syncthreads();
  v8f acc[4] = {};
  gemm_chunk<kKC>(A + (size_t)(m0 + lm) * kC, Bs, lane, acc);

#pragma unroll
  for (int j = 0; j < 4; ++j) {
    const int n = nb + j * 16 + lm;
    const float bv = bias[n];
    const int which = n / kC, rem = n % kC, head = rem >> 5, d = rem & 31;
#pragma unroll
    for (int r = 0; r < 8; ++r) {
      const int gm = m0 + r + half * 8;
      const int w = gm / 49, t = gm % 49;
      const __bf16 val = (__bf16)(acc[j][r] + bv);
      if (which == 0)
        qb[(((size_t)w * kNH + head) * kNP + t) * kHD + d] = val;
      else if (which == 1)
        kb[(((size_t)w * kNH + head) * kNP + t) * kHD + d] = val;
      else
        vt[(((size_t)w * kNH + head) * kHD + d) * kNP + t] = val;
    }
  }
}

// ---------------------------------------------------------------------------
// Kernel 4: per (window, head) attention. S = QK^T/sqrt(32) (16 WMMA tiles),
// masked softmax over 49 keys in LDS, O = P.V (8 WMMA tiles, A from LDS).
__global__ void attn_win(const __bf16* __restrict__ qb, const __bf16* __restrict__ kb,
                         const __bf16* __restrict__ vt, float* __restrict__ out) {
  __shared__ float  S[kNP][kNP + 4];
  __shared__ __bf16 P[kNP][kNP];
  const int wh = blockIdx.x;             // window*6 + head
  const int w = wh / kNH, h = wh % kNH;
  const int lane = threadIdx.x & 31, wv = threadIdx.x >> 5;
  const int half = lane >> 4, lm = lane & 15;
  const __bf16* q = qb + (size_t)wh * kNP * kHD;
  const __bf16* k = kb + (size_t)wh * kNP * kHD;
  const __bf16* v = vt + (size_t)wh * kHD * kNP;

#pragma unroll
  for (int sp = 0; sp < 2; ++sp) {
    const int s = wv + sp * 8;
    const int m0 = (s >> 2) * 16, n0 = (s & 3) * 16;
    v8f acc = wmma_tile<kHD>(q + (size_t)m0 * kHD, kHD,
                             k + (size_t)n0 * kHD, kHD, lane);
#pragma unroll
    for (int r = 0; r < 8; ++r)
      S[m0 + r + half * 8][n0 + lm] = acc[r] * 0.17677669529663687f; // 1/sqrt(32)
  }
  __syncthreads();

  if (threadIdx.x < kNP) {
    const int row = threadIdx.x;
    float mx = -1e30f;
    for (int c = 0; c < 49; ++c) mx = fmaxf(mx, S[row][c]);
    float sum = 0.f;
    for (int c = 0; c < 49; ++c) {
      float e = __expf(S[row][c] - mx);
      S[row][c] = e; sum += e;
    }
    const float inv = 1.f / sum;
    for (int c = 0; c < 49; ++c) P[row][c] = (__bf16)(S[row][c] * inv);
    for (int c = 49; c < kNP; ++c) P[row][c] = (__bf16)0.f;
  }
  __syncthreads();

  {
    const int m0 = (wv >> 1) * 16, n0 = (wv & 1) * 16;
    v8f acc = wmma_tile<kNP>(&P[m0][0], kNP, v + (size_t)n0 * kNP, kNP, lane);
#pragma unroll
    for (int r = 0; r < 8; ++r) {
      const int m = m0 + r + half * 8;
      if (m < 49)
        out[((size_t)w * 49 + m) * kC + h * kHD + n0 + lm] = acc[r];
    }
  }
}

// ---------------------------------------------------------------------------
// Kernel 5: window reverse + shift(+3,+3) gather -> RMSN1 -> +residual -> x1,
// then LN2(x1) -> bf16 MLP A-matrix.
__global__ void stage2(const float* __restrict__ attn, const float* __restrict__ x,
                       const float* __restrict__ sc1, const float* __restrict__ sh1,
                       const float* __restrict__ lg1, const float* __restrict__ lb1,
                       const float* __restrict__ g2,  const float* __restrict__ b2,
                       float* __restrict__ x1, __bf16* __restrict__ Amlp) {
  const int p = (int)((blockIdx.x * blockDim.x + threadIdx.x) >> 5);
  const int lane = threadIdx.x & 31;
  if (p >= kM) return;
  const int b = p / kL, l = p % kL;
  const int hh = l / kHW, wp = l % kHW;
  const int hs = (hh + kHW - kSH) % kHW, vs = (wp + kHW - kSH) % kHW;
  const int wwin = (b * 8 + hs / 7) * 8 + vs / 7;
  const int t = (hs % 7) * 7 + (vs % 7);
  const float* hrow = attn + ((size_t)wwin * 49 + t) * kC;
  const float* xrow = x + (size_t)p * kC;

  float hv[6], s = 0.f, s2 = 0.f;
#pragma unroll
  for (int kk = 0; kk < 6; ++kk) {
    hv[kk] = hrow[kk * 32 + lane];
    s += hv[kk]; s2 += hv[kk] * hv[kk];
  }
  s = wave_sum(s); s2 = wave_sum(s2);
  const float inv_rms = rsqrtf(s2 * (1.f / kC) + 1e-8f);
  const float meanU = s * (1.f / kC) * inv_rms;
  const float varU  = s2 * (1.f / kC) * inv_rms * inv_rms - meanU * meanU;
  const float invs  = rsqrtf(varU + 1e-5f);

  float xv[6], t1 = 0.f, t2 = 0.f;
  float* x1row = x1 + (size_t)p * kC;
#pragma unroll
  for (int kk = 0; kk < 6; ++kk) {
    int c = kk * 32 + lane;
    float ln = (hv[kk] * inv_rms - meanU) * invs * lg1[c] + lb1[c];
    xv[kk] = xrow[c] + ln * sc1[c] + sh1[c];
    x1row[c] = xv[kk];
    t1 += xv[kk]; t2 += xv[kk] * xv[kk];
  }
  t1 = wave_sum(t1); t2 = wave_sum(t2);
  const float m2 = t1 * (1.f / kC);
  const float i2 = rsqrtf(t2 * (1.f / kC) - m2 * m2 + 1e-5f);
  __bf16* dst = Amlp + (size_t)p * kC;
#pragma unroll
  for (int kk = 0; kk < 6; ++kk) {
    int c = kk * 32 + lane;
    dst[c] = (__bf16)((xv[kk] - m2) * i2 * g2[c] + b2[c]);
  }
}

// ---------------------------------------------------------------------------
// Kernel 6: MLP1 GEMM [50176 x 192] x [192 x 768] + bias + exact GELU -> bf16.
__global__ void mlp1_gemm(const __bf16* __restrict__ A, const __bf16* __restrict__ W,
                          const float* __restrict__ bias, __bf16* __restrict__ H) {
  extern __shared__ __bf16 Bs[];
  const int lane = threadIdx.x & 31, wv = threadIdx.x >> 5;
  const int half = lane >> 4, lm = lane & 15;
  const int m0 = blockIdx.x * 128 + wv * 16;
  const int nb = blockIdx.y * 64;

  stage_B_issue<kKC>(W + (size_t)nb * kC, kC, Bs);
  stage_B_drain();
  __syncthreads();
  v8f acc[4] = {};
  gemm_chunk<kKC>(A + (size_t)(m0 + lm) * kC, Bs, lane, acc);

#pragma unroll
  for (int j = 0; j < 4; ++j) {
    const int n = nb + j * 16 + lm;
    const float bv = bias[n];
#pragma unroll
    for (int r = 0; r < 8; ++r) {
      const int gm = m0 + r + half * 8;
      const float vv = acc[j][r] + bv;
      const float g  = 0.5f * vv * (1.f + erff(vv * 0.70710678118654752f));
      H[(size_t)gm * kHID + n] = (__bf16)g;
    }
  }
}

// ---------------------------------------------------------------------------
// Kernel 7: MLP2 GEMM [50176 x 768] x [768 x 192] + bias + residual -> y.
// K staged in 4 chunks of 192 through two LDS panels: the async DMA of chunk
// i+1 is issued before computing chunk i, so data movement (ASYNCcnt) hides
// under WMMA compute; s_wait_asynccnt 0 + barrier close each stage.
__global__ void mlp2_gemm(const __bf16* __restrict__ A, const __bf16* __restrict__ W,
                          const float* __restrict__ bias, const float* __restrict__ x1,
                          float* __restrict__ y) {
  extern __shared__ __bf16 Bs[];
  __bf16* panel0 = Bs;
  __bf16* panel1 = Bs + kPanel;
  const int lane = threadIdx.x & 31, wv = threadIdx.x >> 5;
  const int half = lane >> 4, lm = lane & 15;
  const int m0 = blockIdx.x * 128 + wv * 16;
  const int nb = blockIdx.y * 64;
  const __bf16* Wbase = W + (size_t)nb * kHID;
  const __bf16* arow  = A + (size_t)(m0 + lm) * kHID;

  v8f acc[4] = {};
  stage_B_issue<kKC>(Wbase, kHID, panel0);
  stage_B_drain();
  __syncthreads();
#pragma unroll
  for (int i = 0; i < kHID / kKC; ++i) {          // 4 chunks
    __bf16* cur = (i & 1) ? panel1 : panel0;
    __bf16* nxt = (i & 1) ? panel0 : panel1;
    if (i < kHID / kKC - 1)
      stage_B_issue<kKC>(Wbase + (i + 1) * kKC, kHID, nxt);
    gemm_chunk<kKC>(arow + i * kKC, cur, lane, acc);
    if (i < kHID / kKC - 1) {
      stage_B_drain();
      __syncthreads();
    }
  }

#pragma unroll
  for (int j = 0; j < 4; ++j) {
    const int n = nb + j * 16 + lm;
    const float bv = bias[n];
#pragma unroll
    for (int r = 0; r < 8; ++r) {
      const int gm = m0 + r + half * 8;
      y[(size_t)gm * kC + n] = acc[j][r] + bv + x1[(size_t)gm * kC + n];
    }
  }
}

// ---------------------------------------------------------------------------
// Kernel 8: final RMSN2 -> output (fp32).
__global__ void stage3(const float* __restrict__ y,
                       const float* __restrict__ sc, const float* __restrict__ sh,
                       const float* __restrict__ g,  const float* __restrict__ b,
                       float* __restrict__ out) {
  const int p = (int)((blockIdx.x * blockDim.x + threadIdx.x) >> 5);
  const int lane = threadIdx.x & 31;
  if (p >= kM) return;
  const float* row = y + (size_t)p * kC;
  float v[6], s = 0.f, s2 = 0.f;
#pragma unroll
  for (int kk = 0; kk < 6; ++kk) {
    v[kk] = row[kk * 32 + lane];
    s += v[kk]; s2 += v[kk] * v[kk];
  }
  s = wave_sum(s); s2 = wave_sum(s2);
  const float inv_rms = rsqrtf(s2 * (1.f / kC) + 1e-8f);
  const float meanU = s * (1.f / kC) * inv_rms;
  const float varU  = s2 * (1.f / kC) * inv_rms * inv_rms - meanU * meanU;
  const float invs  = rsqrtf(varU + 1e-5f);
  float* dst = out + (size_t)p * kC;
#pragma unroll
  for (int kk = 0; kk < 6; ++kk) {
    int c = kk * 32 + lane;
    float ln = (v[kk] * inv_rms - meanU) * invs * g[c] + b[c];
    dst[c] = ln * sc[c] + sh[c];
  }
}

// ---------------------------------------------------------------------------
extern "C" void kernel_launch(void* const* d_in, const int* in_sizes, int n_in,
                              void* d_out, int out_size, void* d_ws, size_t ws_size,
                              hipStream_t stream) {
  (void)in_sizes; (void)n_in; (void)out_size; (void)ws_size;
  const float* x         = (const float*)d_in[0];
  const float* norm1_g   = (const float*)d_in[1];
  const float* norm1_b   = (const float*)d_in[2];
  const float* attn_ln_g = (const float*)d_in[3];
  const float* attn_ln_b = (const float*)d_in[4];
  const float* qkv_w     = (const float*)d_in[5];
  const float* qkv_b     = (const float*)d_in[6];
  const float* r1_scale  = (const float*)d_in[7];
  const float* r1_shift  = (const float*)d_in[8];
  const float* r1_g      = (const float*)d_in[9];
  const float* r1_b      = (const float*)d_in[10];
  const float* norm2_g   = (const float*)d_in[11];
  const float* norm2_b   = (const float*)d_in[12];
  const float* mlp_w1    = (const float*)d_in[13];
  const float* mlp_b1    = (const float*)d_in[14];
  const float* mlp_w2    = (const float*)d_in[15];
  const float* mlp_b2    = (const float*)d_in[16];
  const float* r2_scale  = (const float*)d_in[17];
  const float* r2_shift  = (const float*)d_in[18];
  const float* r2_g      = (const float*)d_in[19];
  const float* r2_b      = (const float*)d_in[20];
  float* out = (float*)d_out;

  char* ws = (char*)d_ws;
  size_t off = 0;
  auto take = [&](size_t bytes) -> void* {
    void* p = ws + off;
    off = (off + bytes + 255) & ~(size_t)255;
    return p;
  };
  const int nWin = kB * (kHW / 7) * (kHW / 7);                 // 1024
  __bf16* wq    = (__bf16*)take((size_t)3 * kC * kC * 2);
  __bf16* w1    = (__bf16*)take((size_t)kHID * kC * 2);
  __bf16* w2    = (__bf16*)take((size_t)kC * kHID * 2);
  __bf16* Awin  = (__bf16*)take((size_t)kM * kC * 2);          // reused as Amlp
  __bf16* qb    = (__bf16*)take((size_t)nWin * kNH * kNP * kHD * 2);
  __bf16* kb    = (__bf16*)take((size_t)nWin * kNH * kNP * kHD * 2);
  __bf16* vt    = (__bf16*)take((size_t)nWin * kNH * kHD * kNP * 2);
  float*  attno = (float*)take((size_t)kM * kC * 4);
  float*  x1    = (float*)take((size_t)kM * kC * 4);
  float*  ybuf  = (float*)take((size_t)kM * kC * 4);
  __bf16* Amlp  = Awin;                         // Awin dead after qkv_gemm
  __bf16* Hbuf  = qb;                           // q/k/vt/attn dead after stage2

  const size_t qkvBytes = (size_t)3 * nWin * kNH * kNP * kHD * 2;
  (void)hipMemsetAsync(qb, 0, qkvBytes, stream); // zero padded K/V rows

  const size_t panelLDS  = (size_t)kPanel * 2;                 // 25.6 KB
  const size_t panelLDS2 = panelLDS * 2;                       // 51.2 KB

  prep_weights<<<(kHID * kC + 255) / 256, 256, 0, stream>>>(
      qkv_w, mlp_w1, mlp_w2, wq, w1, w2);

  stage1_ln_window<<<kM / 8, 256, 0, stream>>>(
      x, norm1_g, norm1_b, attn_ln_g, attn_ln_b, Awin);

  qkv_gemm<<<dim3(kM / 128, (3 * kC) / 64), 256, panelLDS, stream>>>(
      Awin, wq, qkv_b, qb, kb, vt);

  attn_win<<<nWin * kNH, 256, 0, stream>>>(qb, kb, vt, attno);

  stage2<<<kM / 8, 256, 0, stream>>>(
      attno, x, r1_scale, r1_shift, r1_g, r1_b, norm2_g, norm2_b, x1, Amlp);

  mlp1_gemm<<<dim3(kM / 128, kHID / 64), 256, panelLDS, stream>>>(
      Amlp, w1, mlp_b1, Hbuf);

  mlp2_gemm<<<dim3(kM / 128, kC / 64), 256, panelLDS2, stream>>>(
      Hbuf, w2, mlp_b2, x1, ybuf);

  stage3<<<kM / 8, 256, 0, stream>>>(ybuf, r2_scale, r2_shift, r2_g, r2_b, out);
}
